// MultiHeadSelfAttention_14740327760466
// MI455X (gfx1250) — matrix operand order repaired and checked by reading, not compiled, not run on hardware
//
#include <hip/hip_runtime.h>

// B=4, T=2048, C=1024, H=16, hd=64
// d_in: sequences f32 [4,2048,1024], w_attn f32 [1024,3072], b_attn f32 [3072],
//       w_proj f32 [1024,1024], b_proj f32 [1024]
// d_out: f32 [4,2048,1024]

typedef __attribute__((ext_vector_type(16))) __bf16 bf16x16;
typedef __attribute__((ext_vector_type(8)))  float  f32x8;

union Frag {          // one WMMA 16-bit A/B fragment: 16 bf16 = 32 B = 8 VGPRs
  bf16x16 v;
  uint4   u[2];
};
union BF4 { __bf16 e[4]; uint2 u; };
union BF8 { __bf16 e[8]; uint4 u; };

__device__ __forceinline__ __bf16 f2bf(float f) {
  unsigned u = __builtin_bit_cast(unsigned, f);
  u += 0x7fffu + ((u >> 16) & 1u);            // round-to-nearest-even
  unsigned short s = (unsigned short)(u >> 16);
  return __builtin_bit_cast(__bf16, s);
}

__device__ __forceinline__ f32x8 zero8() {
  f32x8 z = {0.f, 0.f, 0.f, 0.f, 0.f, 0.f, 0.f, 0.f};
  return z;
}

__device__ __forceinline__ f32x8 wmma_bf16(bf16x16 a, bf16x16 b, f32x8 c) {
  return __builtin_amdgcn_wmma_f32_16x16x32_bf16(false, a, false, b, (short)0, c,
                                                 false, false);
}

// flat LDS pointer -> 32-bit LDS byte address (aperture low bits)
__device__ __forceinline__ unsigned ldsoff(const void* p) {
  return (unsigned)(unsigned long long)p;
}

// CDNA5 async copy: 16B global -> LDS, tracked with ASYNCcnt (no VGPR data)
__device__ __forceinline__ void async_ld_b128(unsigned lds, const void* g) {
  asm volatile("global_load_async_to_lds_b128 %0, %1, off"
               :
               : "v"(lds), "v"(g)
               : "memory");
}
__device__ __forceinline__ void wait_async0() {
  asm volatile("s_wait_asynccnt 0" ::: "memory");
}

// ---------------------------------------------------------------------------
// Prep kernel A: f32 -> bf16 straight copy (for X)
// ---------------------------------------------------------------------------
__global__ __launch_bounds__(256) void cvt_bf16_kernel(
    const float* __restrict__ in, __bf16* __restrict__ out, int n8) {
  const int i = blockIdx.x * 256 + threadIdx.x;
  if (i >= n8) return;
  const float4 a = *reinterpret_cast<const float4*>(in + (size_t)i * 8);
  const float4 b = *reinterpret_cast<const float4*>(in + (size_t)i * 8 + 4);
  BF8 pk;
  pk.e[0] = f2bf(a.x); pk.e[1] = f2bf(a.y); pk.e[2] = f2bf(a.z); pk.e[3] = f2bf(a.w);
  pk.e[4] = f2bf(b.x); pk.e[5] = f2bf(b.y); pk.e[6] = f2bf(b.z); pk.e[7] = f2bf(b.w);
  *reinterpret_cast<uint4*>(out + (size_t)i * 8) = pk.u;
}

// ---------------------------------------------------------------------------
// Prep kernel B: W[K=1024][N] f32 -> Wt[N][1024] bf16 (LDS-tiled transpose)
// ---------------------------------------------------------------------------
__global__ __launch_bounds__(256) void transpose_cvt_kernel(
    const float* __restrict__ W, __bf16* __restrict__ Wt, int N) {
  __shared__ float Tl[64][65];
  const int tid = threadIdx.x;
  const int n0 = blockIdx.x * 64, k0 = blockIdx.y * 64;
  {
    const int r = tid >> 4, cg = (tid & 15) * 4;
#pragma unroll
    for (int p = 0; p < 4; ++p) {
      const float4 v = *reinterpret_cast<const float4*>(
          W + (size_t)(k0 + r + p * 16) * N + n0 + cg);
      Tl[r + p * 16][cg + 0] = v.x;
      Tl[r + p * 16][cg + 1] = v.y;
      Tl[r + p * 16][cg + 2] = v.z;
      Tl[r + p * 16][cg + 3] = v.w;
    }
  }
  __syncthreads();
  {
    const int nl = tid >> 4, kg = (tid & 15) * 4;
#pragma unroll
    for (int p = 0; p < 4; ++p) {
      BF4 pk;
#pragma unroll
      for (int j = 0; j < 4; ++j) pk.e[j] = f2bf(Tl[kg + j][nl + p * 16]);
      *reinterpret_cast<uint2*>(Wt + (size_t)(n0 + nl + p * 16) * 1024 + k0 + kg) =
          pk.u;
    }
  }
}

// ---------------------------------------------------------------------------
// Kernel 1: QKV GEMM  Xbf[8192,1024] @ WtT  -> Q,K bf16 [B,H,T,64], V bf16 [B,H,64,T]
// 128x128 block tile, BK=32, 2x4 wave grid, double-buffered async LDS staging.
// ---------------------------------------------------------------------------
__global__ __launch_bounds__(256) void qkv_gemm_kernel(
    const __bf16* __restrict__ X,     // [8192,1024] bf16
    const __bf16* __restrict__ Wt,    // [3072,1024] bf16 (pre-transposed)
    const float* __restrict__ bias,   // [3072]
    __bf16* __restrict__ q_ws, __bf16* __restrict__ k_ws,
    __bf16* __restrict__ v_ws) {
  __shared__ __bf16 Alds[2][128][40];   // [m][k], 80B row stride (16B aligned)
  __shared__ __bf16 Blds[2][128][40];   // [n][k]

  const int tid  = threadIdx.x;
  const int wave = tid >> 5, lane = tid & 31;
  const int lh = lane >> 4, lr = lane & 15;
  const int wm = wave >> 2, wn = wave & 3;
  const int m0 = blockIdx.y * 128, n0 = blockIdx.x * 128;

  // copy geometry: each thread owns 32B (two b128) of each tile
  const int cr = tid >> 1, cc = (tid & 1) * 16;
  const __bf16* gA = X + (size_t)(m0 + cr) * 1024 + cc;
  const __bf16* gB = Wt + (size_t)(n0 + cr) * 1024 + cc;

  f32x8 acc[4][2];
#pragma unroll
  for (int im = 0; im < 4; ++im)
#pragma unroll
    for (int in = 0; in < 2; ++in) acc[im][in] = zero8();

  // prologue: async-stage tile 0 into buffer 0
  async_ld_b128(ldsoff(&Alds[0][cr][cc]), gA);
  async_ld_b128(ldsoff(&Alds[0][cr][cc + 8]), gA + 8);
  async_ld_b128(ldsoff(&Blds[0][cr][cc]), gB);
  async_ld_b128(ldsoff(&Blds[0][cr][cc + 8]), gB + 8);

  for (int it = 0; it < 32; ++it) {
    const int cur = it & 1;
    wait_async0();
    __syncthreads();   // tile `cur` resident; everyone done reading `1-cur`
    if (it + 1 < 32) { // overlap next tile's copy with this tile's WMMAs
      const int kk = (it + 1) * 32;
      async_ld_b128(ldsoff(&Alds[1 - cur][cr][cc]), gA + kk);
      async_ld_b128(ldsoff(&Alds[1 - cur][cr][cc + 8]), gA + kk + 8);
      async_ld_b128(ldsoff(&Blds[1 - cur][cr][cc]), gB + kk);
      async_ld_b128(ldsoff(&Blds[1 - cur][cr][cc + 8]), gB + kk + 8);
    }

    Frag afr[4], bfr[2];
#pragma unroll
    for (int im = 0; im < 4; ++im) {  // A 16x32: chunks at 8*half, 16+8*half
      const __bf16* rp = &Alds[cur][wm * 64 + im * 16 + lr][0];
      afr[im].u[0] = *reinterpret_cast<const uint4*>(rp + 8 * lh);
      afr[im].u[1] = *reinterpret_cast<const uint4*>(rp + 16 + 8 * lh);
    }
#pragma unroll
    for (int in = 0; in < 2; ++in) {  // B 32x16: chunks at 8*half, 16+8*half
      const __bf16* rp = &Blds[cur][wn * 32 + in * 16 + lr][0];
      bfr[in].u[0] = *reinterpret_cast<const uint4*>(rp + 8 * lh);
      bfr[in].u[1] = *reinterpret_cast<const uint4*>(rp + 16 + 8 * lh);
    }
#pragma unroll
    for (int im = 0; im < 4; ++im)
#pragma unroll
      for (int in = 0; in < 2; ++in)
        acc[im][in] = wmma_bf16(afr[im].v, bfr[in].v, acc[im][in]);
  }

  // Epilogue: +bias, split q/k/v.  q,k: [B,H,T,64].  v: [B,H,64,T] (transposed)
#pragma unroll
  for (int im = 0; im < 4; ++im) {
#pragma unroll
    for (int in = 0; in < 2; ++in) {
      const int ng = n0 + wn * 32 + in * 16 + lr;
      const int which = ng >> 10;  // uniform within a 16-wide tile
      const int c = ng & 1023, h = c >> 6, d = c & 63;
      const float bv = bias[ng];
#pragma unroll
      for (int j = 0; j < 8; ++j) {  // C layout: row = 8*half + j, col = lr
        const int mg = m0 + wm * 64 + im * 16 + lh * 8 + j;
        const int bb = mg >> 11, tt = mg & 2047;
        const __bf16 val = f2bf(acc[im][in][j] + bv);
        if (which == 0)
          q_ws[((size_t)(bb * 16 + h) * 2048 + tt) * 64 + d] = val;
        else if (which == 1)
          k_ws[((size_t)(bb * 16 + h) * 2048 + tt) * 64 + d] = val;
        else
          v_ws[((size_t)(bb * 16 + h) * 64 + d) * 2048 + tt] = val;
      }
    }
  }
}

// ---------------------------------------------------------------------------
// Kernel 2: flash attention per (b,h); 8 waves x 16 query rows; 64-key tiles,
// double-buffered async K/V staging (V already [d][key] in global).
// ---------------------------------------------------------------------------
__global__ __launch_bounds__(256) void attn_fwd_kernel(
    const __bf16* __restrict__ q_ws, const __bf16* __restrict__ k_ws,
    const __bf16* __restrict__ v_ws, __bf16* __restrict__ y_ws) {
  __shared__ __bf16 Klds[2][64][72];   // [key][d]  144B row stride
  __shared__ __bf16 Vlds[2][64][72];   // [d][key]
  __shared__ __bf16 Plds[8][16][72];   // per-wave P staging [row][key]

  const int tid  = threadIdx.x;
  const int wave = tid >> 5, lane = tid & 31;
  const int lh = lane >> 4, lr = lane & 15;
  const int bh = blockIdx.y;
  const int q0 = blockIdx.x * 128 + wave * 16;
  const size_t kbase = (size_t)bh * 2048 * 64;  // K: [T][64]
  const size_t vbase = (size_t)bh * 64 * 2048;  // V: [64][T]

  // copy geometry: rows crr/crr+32, 16B column chunk cc8
  const int crr = tid >> 3, cc8 = (tid & 7) * 8;
  const __bf16* gK = k_ws + kbase + (size_t)crr * 64 + cc8;
  const __bf16* gV = v_ws + vbase + (size_t)crr * 2048 + cc8;

  Frag qf[2];  // resident Q: 16x64 as two 16x32 A-fragments
  {
    const __bf16* qrow = q_ws + kbase + (size_t)(q0 + lr) * 64;
#pragma unroll
    for (int ks = 0; ks < 2; ++ks) {
      qf[ks].u[0] = *reinterpret_cast<const uint4*>(qrow + ks * 32 + 8 * lh);
      qf[ks].u[1] = *reinterpret_cast<const uint4*>(qrow + ks * 32 + 16 + 8 * lh);
    }
  }

  f32x8 O[4];
#pragma unroll
  for (int dt = 0; dt < 4; ++dt) O[dt] = zero8();
  float mrow[8], lsum[8];
#pragma unroll
  for (int j = 0; j < 8; ++j) { mrow[j] = -3.0e38f; lsum[j] = 0.0f; }

  // prologue: stage key-tile 0
  async_ld_b128(ldsoff(&Klds[0][crr][cc8]), gK);
  async_ld_b128(ldsoff(&Klds[0][crr + 32][cc8]), gK + 32 * 64);
  async_ld_b128(ldsoff(&Vlds[0][crr][cc8]), gV);
  async_ld_b128(ldsoff(&Vlds[0][crr + 32][cc8]), gV + (size_t)32 * 2048);

  for (int it = 0; it < 32; ++it) {
    const int cur = it & 1;
    wait_async0();
    __syncthreads();
    if (it + 1 < 32) {
      const int kt = (it + 1) * 64;
      async_ld_b128(ldsoff(&Klds[1 - cur][crr][cc8]), gK + (size_t)kt * 64);
      async_ld_b128(ldsoff(&Klds[1 - cur][crr + 32][cc8]),
                    gK + (size_t)(kt + 32) * 64);
      async_ld_b128(ldsoff(&Vlds[1 - cur][crr][cc8]), gV + kt);
      async_ld_b128(ldsoff(&Vlds[1 - cur][crr + 32][cc8]),
                    gV + (size_t)32 * 2048 + kt);
    }

    f32x8 S[4];  // 16 queries x 64 keys
#pragma unroll
    for (int nt = 0; nt < 4; ++nt) {
      S[nt] = zero8();
#pragma unroll
      for (int ks = 0; ks < 2; ++ks) {
        Frag kf;  // B frag: col = key, K-dim = d (contiguous in Klds)
        const __bf16* rp = &Klds[cur][nt * 16 + lr][0];
        kf.u[0] = *reinterpret_cast<const uint4*>(rp + ks * 32 + 8 * lh);
        kf.u[1] = *reinterpret_cast<const uint4*>(rp + ks * 32 + 16 + 8 * lh);
        S[nt] = wmma_bf16(qf[ks].v, kf.v, S[nt]);
      }
    }

    float pv[4][8];
#pragma unroll
    for (int j = 0; j < 8; ++j) {  // row = 8*lh + j spans the 16 lanes of a half
      float mx = -3.0e38f;
#pragma unroll
      for (int nt = 0; nt < 4; ++nt) {
        S[nt][j] *= 0.125f;  // 1/sqrt(64)
        mx = fmaxf(mx, S[nt][j]);
      }
#pragma unroll
      for (int m = 1; m < 16; m <<= 1) mx = fmaxf(mx, __shfl_xor(mx, m, 32));
      const float mnew = fmaxf(mrow[j], mx);
      const float corr = __expf(mrow[j] - mnew);
      float rs = 0.0f;
#pragma unroll
      for (int nt = 0; nt < 4; ++nt) {
        const float p = __expf(S[nt][j] - mnew);
        pv[nt][j] = p;
        rs += p;
      }
#pragma unroll
      for (int m = 1; m < 16; m <<= 1) rs += __shfl_xor(rs, m, 32);
      lsum[j] = lsum[j] * corr + rs;
      mrow[j] = mnew;
#pragma unroll
      for (int dt = 0; dt < 4; ++dt) O[dt][j] *= corr;
    }

    // C-layout P -> per-wave LDS -> A-fragment layout (same-wave LDS in order)
#pragma unroll
    for (int nt = 0; nt < 4; ++nt)
#pragma unroll
      for (int j = 0; j < 8; ++j)
        Plds[wave][lh * 8 + j][nt * 16 + lr] = f2bf(pv[nt][j]);

    Frag pf[2];
#pragma unroll
    for (int ks = 0; ks < 2; ++ks) {
      const __bf16* rp = &Plds[wave][lr][0];
      pf[ks].u[0] = *reinterpret_cast<const uint4*>(rp + ks * 32 + 8 * lh);
      pf[ks].u[1] = *reinterpret_cast<const uint4*>(rp + ks * 32 + 16 + 8 * lh);
    }
#pragma unroll
    for (int dt = 0; dt < 4; ++dt) {
#pragma unroll
      for (int ks = 0; ks < 2; ++ks) {
        Frag vf;  // B frag: col = d, K-dim = key (contiguous in Vlds)
        const __bf16* rp = &Vlds[cur][dt * 16 + lr][0];
        vf.u[0] = *reinterpret_cast<const uint4*>(rp + ks * 32 + 8 * lh);
        vf.u[1] = *reinterpret_cast<const uint4*>(rp + ks * 32 + 16 + 8 * lh);
        O[dt] = wmma_bf16(pf[ks].v, vf.v, O[dt]);
      }
    }
  }

  const int b = bh >> 4, h = bh & 15;
#pragma unroll
  for (int j = 0; j < 8; ++j) {
    const float inv = 1.0f / lsum[j];
    const int trow = q0 + lh * 8 + j;
#pragma unroll
    for (int dt = 0; dt < 4; ++dt)
      y_ws[(size_t)(b * 2048 + trow) * 1024 + h * 64 + dt * 16 + lr] =
          f2bf(O[dt][j] * inv);
  }
}

// ---------------------------------------------------------------------------
// Kernel 3: projection  Y[8192,1024]bf16 @ WpT[1024,1024]bf16 + bp -> out f32
// ---------------------------------------------------------------------------
__global__ __launch_bounds__(256) void proj_gemm_kernel(
    const __bf16* __restrict__ Y, const __bf16* __restrict__ Wt,
    const float* __restrict__ bias, float* __restrict__ out) {
  __shared__ __bf16 Alds[2][128][40];
  __shared__ __bf16 Blds[2][128][40];

  const int tid  = threadIdx.x;
  const int wave = tid >> 5, lane = tid & 31;
  const int lh = lane >> 4, lr = lane & 15;
  const int wm = wave >> 2, wn = wave & 3;
  const int m0 = blockIdx.y * 128, n0 = blockIdx.x * 128;

  const int cr = tid >> 1, cc = (tid & 1) * 16;
  const __bf16* gA = Y + (size_t)(m0 + cr) * 1024 + cc;
  const __bf16* gB = Wt + (size_t)(n0 + cr) * 1024 + cc;

  f32x8 acc[4][2];
#pragma unroll
  for (int im = 0; im < 4; ++im)
#pragma unroll
    for (int in = 0; in < 2; ++in) acc[im][in] = zero8();

  async_ld_b128(ldsoff(&Alds[0][cr][cc]), gA);
  async_ld_b128(ldsoff(&Alds[0][cr][cc + 8]), gA + 8);
  async_ld_b128(ldsoff(&Blds[0][cr][cc]), gB);
  async_ld_b128(ldsoff(&Blds[0][cr][cc + 8]), gB + 8);

  for (int it = 0; it < 32; ++it) {
    const int cur = it & 1;
    wait_async0();
    __syncthreads();
    if (it + 1 < 32) {
      const int kk = (it + 1) * 32;
      async_ld_b128(ldsoff(&Alds[1 - cur][cr][cc]), gA + kk);
      async_ld_b128(ldsoff(&Alds[1 - cur][cr][cc + 8]), gA + kk + 8);
      async_ld_b128(ldsoff(&Blds[1 - cur][cr][cc]), gB + kk);
      async_ld_b128(ldsoff(&Blds[1 - cur][cr][cc + 8]), gB + kk + 8);
    }

    Frag afr[4], bfr[2];
#pragma unroll
    for (int im = 0; im < 4; ++im) {
      const __bf16* rp = &Alds[cur][wm * 64 + im * 16 + lr][0];
      afr[im].u[0] = *reinterpret_cast<const uint4*>(rp + 8 * lh);
      afr[im].u[1] = *reinterpret_cast<const uint4*>(rp + 16 + 8 * lh);
    }
#pragma unroll
    for (int in = 0; in < 2; ++in) {
      const __bf16* rp = &Blds[cur][wn * 32 + in * 16 + lr][0];
      bfr[in].u[0] = *reinterpret_cast<const uint4*>(rp + 8 * lh);
      bfr[in].u[1] = *reinterpret_cast<const uint4*>(rp + 16 + 8 * lh);
    }
#pragma unroll
    for (int im = 0; im < 4; ++im)
#pragma unroll
      for (int in = 0; in < 2; ++in)
        acc[im][in] = wmma_bf16(afr[im].v, bfr[in].v, acc[im][in]);
  }

#pragma unroll
  for (int im = 0; im < 4; ++im) {
#pragma unroll
    for (int in = 0; in < 2; ++in) {
      const int ng = n0 + wn * 32 + in * 16 + lr;
      const float bv = bias[ng];
#pragma unroll
      for (int j = 0; j < 8; ++j) {
        const int mg = m0 + wm * 64 + im * 16 + lh * 8 + j;
        out[(size_t)mg * 1024 + ng] = acc[im][in][j] + bv;
      }
    }
  }
}

// ---------------------------------------------------------------------------
extern "C" void kernel_launch(void* const* d_in, const int* in_sizes, int n_in,
                              void* d_out, int out_size, void* d_ws,
                              size_t ws_size, hipStream_t stream) {
  const float* seq    = (const float*)d_in[0];
  const float* w_attn = (const float*)d_in[1];
  const float* b_attn = (const float*)d_in[2];
  const float* w_proj = (const float*)d_in[3];
  const float* b_proj = (const float*)d_in[4];
  float* out = (float*)d_out;

  char* ws = (char*)d_ws;
  const size_t MiB = (size_t)1024 * 1024;
  __bf16* xbf  = (__bf16*)(ws);                 // 16 MiB  [8192,1024]
  __bf16* watT = (__bf16*)(ws + 16 * MiB);      //  6 MiB  [3072,1024]
  __bf16* wprT = (__bf16*)(ws + 22 * MiB);      //  2 MiB  [1024,1024]
  __bf16* q_ws = (__bf16*)(ws + 24 * MiB);      // 16 MiB  [B,H,T,64]
  __bf16* k_ws = (__bf16*)(ws + 40 * MiB);      // 16 MiB  [B,H,T,64]
  __bf16* v_ws = (__bf16*)(ws + 56 * MiB);      // 16 MiB  [B,H,64,T]
  __bf16* y_ws = (__bf16*)(ws + 72 * MiB);      // 16 MiB  [B,T,C]

  // prep: X -> bf16; weights -> transposed bf16
  cvt_bf16_kernel<<<dim3(4096), 256, 0, stream>>>(seq, xbf, 1048576);
  transpose_cvt_kernel<<<dim3(48, 16), 256, 0, stream>>>(w_attn, watT, 3072);
  transpose_cvt_kernel<<<dim3(16, 16), 256, 0, stream>>>(w_proj, wprT, 1024);

  // M=8192, N=3072
  qkv_gemm_kernel<<<dim3(24, 64), 256, 0, stream>>>(xbf, watT, b_attn, q_ws,
                                                    k_ws, v_ws);
  // (T/128, B*H)
  attn_fwd_kernel<<<dim3(16, 64), 256, 0, stream>>>(q_ws, k_ws, v_ws, y_ws);
  // M=8192, N=1024
  proj_gemm_kernel<<<dim3(8, 64), 256, 0, stream>>>(y_ws, wprT, b_proj, out);
}
